// SwinTransformerBlock_37546604101789
// MI455X (gfx1250) — matrix operand order repaired and checked by reading, not compiled, not run on hardware
//
#include <hip/hip_runtime.h>
#include <hip/hip_bf16.h>

// ---------------------------------------------------------------------------
// Types for CDNA5 WMMA (wave32, 16x16x32 bf16 -> f32)
// ---------------------------------------------------------------------------
typedef __bf16 bf16;
typedef __attribute__((ext_vector_type(16))) __bf16 v16bf;
typedef __attribute__((ext_vector_type(8)))  __bf16 v8bf;
typedef __attribute__((ext_vector_type(8)))  float  v8f;

#define WARP 32

__device__ __forceinline__ v8f wmma_bf16(v16bf a, v16bf b, v8f c) {
  return __builtin_amdgcn_wmma_f32_16x16x32_bf16(
      false, a, false, b, (short)0, c, false, false);
}

__device__ __forceinline__ v16bf cat8(v8bf lo, v8bf hi) {
  return __builtin_shufflevector(lo, hi, 0, 1, 2, 3, 4, 5, 6, 7,
                                 8, 9, 10, 11, 12, 13, 14, 15);
}

// ds_swizzle_b32 xor-lane reductions (imm = xor<<10 | and=0x1F, group-of-32)
template <int IMM>
__device__ __forceinline__ float swz_add(float v) {
  return v + __int_as_float(__builtin_amdgcn_ds_swizzle(__float_as_int(v), IMM));
}
template <int IMM>
__device__ __forceinline__ float swz_max(float v) {
  return fmaxf(v, __int_as_float(__builtin_amdgcn_ds_swizzle(__float_as_int(v), IMM)));
}
#define SWZ_X1  0x041F
#define SWZ_X2  0x081F
#define SWZ_X4  0x101F
#define SWZ_X8  0x201F
#define SWZ_X16 0x401F

// A fragment: 16x32 bf16 row-major, ld multiple of 8 (16B-aligned rows).
__device__ __forceinline__ v16bf load_a_frag(const bf16* base, int ld,
                                             int m0, int k0, int lane) {
  const int half = (lane >> 4) & 1;
  const int m = m0 + (lane & 15);
  const bf16* row = base + (size_t)m * ld + k0 + half * 8;
  v8bf lo = *(const v8bf*)(row);        // K = half*8 + 0..7
  v8bf hi = *(const v8bf*)(row + 16);   // K = 16 + half*8 + 0..7
  return cat8(lo, hi);
}

// B fragment from transposed layout Bt[n][k]: K = k0 + half*8 + e (e<8), k0 + 16 + half*8 + (e-8) (e>=8).
__device__ __forceinline__ v16bf load_bt_frag(const bf16* base, int ld,
                                              int k0, int n0, int lane) {
  const int half = (lane >> 4) & 1;
  const int n = n0 + (lane & 15);
  const bf16* row = base + (size_t)n * ld + k0 + half * 8;
  v8bf lo = *(const v8bf*)(row);
  v8bf hi = *(const v8bf*)(row + 16);
  return cat8(lo, hi);
}

// ---------------------------------------------------------------------------
// Problem constants
// ---------------------------------------------------------------------------
#define Bq     8
#define Hq     56
#define Wq     56
#define Cq     384
#define HEADS  12
#define HD     32
#define WS     7
#define SS     3
#define Nq     49        // window tokens
#define NP     64        // padded window tokens (attention only)
#define NWIN   512
#define NWg    64
#define Lq     3136
#define MTOK   (Bq * Lq)       // 25088 tokens (multiple of 256)

// window (win, n) -> token index, including cyclic shift roll(-SS)
__device__ __forceinline__ int win_token(int win, int n) {
  const int b = win >> 6, wl = win & 63;
  const int wh = wl >> 3, ww = wl & 7;
  const int h = (wh * WS + n / WS + SS) % Hq;
  const int w = (ww * WS + n % WS + SS) % Wq;
  return b * Lq + h * Wq + w;
}

// ---------------------------------------------------------------------------
// Weight f32 -> bf16
// ---------------------------------------------------------------------------
__global__ void cvt_bf16(const float* __restrict__ in, bf16* __restrict__ out,
                         int n) {
  int i = blockIdx.x * 256 + threadIdx.x;
  if (i < n) out[i] = (bf16)in[i];
}

// ---------------------------------------------------------------------------
// Row LayerNorm: in f32 [M, C] -> out bf16 (used for LN1 and LN2)
// ---------------------------------------------------------------------------
__global__ __launch_bounds__(128) void ln_rows(
    const float* __restrict__ xin, const float* __restrict__ g,
    const float* __restrict__ bta, bf16* __restrict__ xout) {
  const int row = blockIdx.x;
  const float* px = xin + (size_t)row * Cq;
  bf16* dst = xout + (size_t)row * Cq;
  __shared__ float red[8];
  float xv[3], s = 0.f, s2 = 0.f;
#pragma unroll
  for (int j = 0; j < 3; ++j) {
    int c = threadIdx.x + 128 * j;
    xv[j] = px[c];
    s += xv[j]; s2 += xv[j] * xv[j];
  }
  s = swz_add<SWZ_X1>(s);  s = swz_add<SWZ_X2>(s);  s = swz_add<SWZ_X4>(s);
  s = swz_add<SWZ_X8>(s);  s = swz_add<SWZ_X16>(s);
  s2 = swz_add<SWZ_X1>(s2); s2 = swz_add<SWZ_X2>(s2); s2 = swz_add<SWZ_X4>(s2);
  s2 = swz_add<SWZ_X8>(s2); s2 = swz_add<SWZ_X16>(s2);
  int lane = threadIdx.x & 31, wid = threadIdx.x >> 5;
  if (lane == 0) { red[wid] = s; red[4 + wid] = s2; }
  __syncthreads();
  s  = red[0] + red[1] + red[2] + red[3];
  s2 = red[4] + red[5] + red[6] + red[7];
  const float mu = s * (1.0f / Cq);
  const float rstd = rsqrtf(s2 * (1.0f / Cq) - mu * mu + 1e-6f);
#pragma unroll
  for (int j = 0; j < 3; ++j) {
    int c = threadIdx.x + 128 * j;
    dst[c] = (bf16)((xv[j] - mu) * rstd * g[c] + bta[c]);
  }
}

// ---------------------------------------------------------------------------
// Software-pipelined bf16 GEMM: C[M,N] = A[M,K]*B[K,N]. 256 threads = 8
// waves; 256x64 tile; wave owns 32 rows (2 A-frags x 4 B-frags = 8 WMMA per
// K-step). Register-prefetch pipeline: next tile's global loads issue before
// the current tile's WMMA burst, so VMEM latency overlaps matrix math.
// ---------------------------------------------------------------------------
#define EPI_BF16 0   // +bias, store bf16                    (QKV)
#define EPI_GELU 1   // +bias, exact GELU, store bf16        (FC1)
#define EPI_RES  2   // +bias, +residual, store f32          (proj, FC2)

template <int EPI>
__global__ __launch_bounds__(256) void gemm_bf16(
    const bf16* __restrict__ A, const bf16* __restrict__ Bw,
    const float* __restrict__ bias, int M, int N, int K,
    bf16* __restrict__ out_bf, float* __restrict__ out_f,
    const float* __restrict__ resid) {
  __shared__ __attribute__((aligned(16))) bf16 As[256][40];
  __shared__ __attribute__((aligned(16))) bf16 Bts[64][40];  // Bts[n][k]

  const int tid  = threadIdx.x;
  const int lane = tid & 31;
  const int wid  = tid >> 5;
  const int row0 = blockIdx.y * 256;
  const int col0 = blockIdx.x * 64;

  // fixed per-thread staging slots
  const int rA = tid >> 2;            // A rows rA, rA+64, rA+128, rA+192
  const int cA = (tid & 3) * 8;       // A col chunk
  const int kB = tid >> 3;            // B row (k)
  const int nB = (tid & 7) * 8;       // B col chunk
  const bf16* gA = A + (size_t)(row0 + rA) * K + cA;
  const bf16* gB = Bw + (size_t)kB * N + col0 + nB;

  v8f zero = {};
  v8f acc[2][4] = {{zero, zero, zero, zero}, {zero, zero, zero, zero}};

  v8bf aregs[4], breg;
#pragma unroll
  for (int j = 0; j < 4; ++j)
    aregs[j] = *(const v8bf*)(gA + (size_t)(64 * j) * K);
  breg = *(const v8bf*)(gB);

  const int nk = K >> 5;
  for (int ks = 0; ks < nk; ++ks) {
    // commit staged registers to LDS
#pragma unroll
    for (int j = 0; j < 4; ++j)
      *(v8bf*)&As[rA + 64 * j][cA] = aregs[j];
#pragma unroll
    for (int e = 0; e < 8; ++e) Bts[nB + e][kB] = breg[e];
    __syncthreads();
    // issue next tile's global loads (in flight during the WMMA burst)
    if (ks + 1 < nk) {
      const int k0n = (ks + 1) * 32;
#pragma unroll
      for (int j = 0; j < 4; ++j)
        aregs[j] = *(const v8bf*)(gA + (size_t)(64 * j) * K + k0n);
      breg = *(const v8bf*)(gB + (size_t)k0n * N);
      if (ks + 2 < nk)  // L2 warm-up two tiles ahead (global_prefetch_b8)
        __builtin_prefetch(gA + k0n + 32, 0, 0);
    }
    // compute from LDS
    v16bf a0 = load_a_frag(&As[0][0], 40, wid * 32, 0, lane);
    v16bf a1 = load_a_frag(&As[0][0], 40, wid * 32 + 16, 0, lane);
#pragma unroll
    for (int t = 0; t < 4; ++t) {
      v16bf b = load_bt_frag(&Bts[0][0], 40, 0, t * 16, lane);
      acc[0][t] = wmma_bf16(a0, b, acc[0][t]);
      acc[1][t] = wmma_bf16(a1, b, acc[1][t]);
    }
    __syncthreads();
  }

  const int half = (lane >> 4) & 1;
  const int nlo  = lane & 15;
  float bcol[4];
#pragma unroll
  for (int t = 0; t < 4; ++t) bcol[t] = bias[col0 + t * 16 + nlo];

#pragma unroll
  for (int g = 0; g < 2; ++g) {
#pragma unroll
    for (int r = 0; r < 8; ++r) {
      const int row = row0 + wid * 32 + g * 16 + half * 8 + r;
      bf16*  obrow = (EPI != EPI_RES) ? out_bf + (size_t)row * N + col0 : nullptr;
      float* ofrow = (EPI == EPI_RES) ? out_f + (size_t)row * N + col0 : nullptr;
      const float* rrow = (EPI == EPI_RES) ? resid + (size_t)row * N + col0 : nullptr;
#pragma unroll
      for (int t = 0; t < 4; ++t) {
        const int c = t * 16 + nlo;
        float v = acc[g][t][r] + bcol[t];
        if (EPI == EPI_BF16) {
          obrow[c] = (bf16)v;
        } else if (EPI == EPI_GELU) {
          v = 0.5f * v * (1.0f + erff(v * 0.70710678118f));
          obrow[c] = (bf16)v;
        } else {  // EPI_RES
          ofrow[c] = rrow[c] + v;
        }
      }
    }
  }
}

// ---------------------------------------------------------------------------
// Windowed attention in token layout: one block per (window, head).
// 128 threads = 4 waves. Per-block LDS tables: toks[] (shift-window gather
// map) and codes[] (rel-pos codes) kill all div/mod-7 in the hot paths; the
// combined rpb+mask tile (pad = -1e30) makes softmax input branch-free.
// Reductions use ds_swizzle_b32 (no bpermute address math).
// ---------------------------------------------------------------------------
__global__ __launch_bounds__(128) void attn_window(
    const bf16* __restrict__ qkv, const float* __restrict__ rpb,
    const float* __restrict__ mask, bf16* __restrict__ outp) {
  const int blk = blockIdx.x;
  const int win = blk / HEADS, head = blk % HEADS;
  const int lane = threadIdx.x & 31, wid = threadIdx.x >> 5;

  __shared__ __attribute__((aligned(16))) bf16  qs[NP][40];
  __shared__ __attribute__((aligned(16))) bf16  ks[NP][40];
  __shared__ __attribute__((aligned(16))) bf16  vts[HD][72];  // V^T: [d][key]
  __shared__ __attribute__((aligned(16))) bf16  ps[NP][72];
  __shared__ __attribute__((aligned(16))) float bs[NP][NP];   // rpb+mask tile
  __shared__ int   toks[NP];
  __shared__ short codes[NP];

  const float scale = 0.17677669529663687f;  // 32^-0.5
  const int wl = win & (NWg - 1);

  // ---- per-block tables: token map + rel-pos codes ----
  if (threadIdx.x < NP) {
    const int i = threadIdx.x;
    const int d7 = i / 7, m7 = i - d7 * 7;
    codes[i] = (short)(d7 * (2 * WS - 1) + m7);
    toks[i] = (i < Nq) ? win_token(win, i) : 0;
  }
  __syncthreads();

  // ---- stage q/k/v (gathered through the shifted-window map) ----
  for (int i = threadIdx.x; i < 256; i += 128) {   // 64x32 in v8bf chunks
    const int r = i >> 2, c = (i & 3) * 8;
    if (r < Nq) {
      const size_t off = (size_t)toks[r] * (3 * Cq) + head * HD + c;
      v8bf qv = *(const v8bf*)(qkv + off);
      v8bf kv = *(const v8bf*)(qkv + off + Cq);
      v8bf vv = *(const v8bf*)(qkv + off + 2 * Cq);
      v8bf q8;
#pragma unroll
      for (int e = 0; e < 8; ++e) q8[e] = (bf16)((float)qv[e] * scale);
      *(v8bf*)&qs[r][c] = q8;
      *(v8bf*)&ks[r][c] = kv;
#pragma unroll
      for (int e = 0; e < 8; ++e) vts[c + e][r] = vv[e];
    } else {
      v8bf z8 = {};
      *(v8bf*)&qs[r][c] = z8;
      *(v8bf*)&ks[r][c] = z8;
#pragma unroll
      for (int e = 0; e < 8; ++e) vts[c + e][r] = (bf16)0.0f;
    }
  }
  // ---- build combined bias tile ----
  for (int i = threadIdx.x; i < NP * NP; i += 128) {
    const int row = i >> 6, col = i & 63;
    float b = -1e30f;
    if (row < Nq && col < Nq) {
      const int idx = (int)codes[row] - (int)codes[col]
                      + ((WS - 1) * (2 * WS - 1) + (WS - 1));
      b = rpb[idx * HEADS + head] + mask[((size_t)wl * Nq + row) * Nq + col];
    }
    bs[row][col] = b;
  }
  __syncthreads();

  const int m0 = wid * 16;
  v8f zero = {};
  v8f acc[4] = {zero, zero, zero, zero};
  // S = q @ k^T : ks rows are already the Bt layout for B[d][n] = k[n][d]
  {
    v16bf a = load_a_frag(&qs[0][0], 40, m0, 0, lane);
#pragma unroll
    for (int t = 0; t < 4; ++t) {
      v16bf b = load_bt_frag(&ks[0][0], 40, 0, t * 16, lane);
      acc[t] = wmma_bf16(a, b, acc[t]);
    }
  }

  const int half = (lane >> 4) & 1;
  const int nlo  = lane & 15;
  // branch-free bias add + row softmax (row lives in a 16-lane half)
  float vals[4][8];
#pragma unroll
  for (int r = 0; r < 8; ++r) {
    const int row = m0 + half * 8 + r;
    const float* brow = &bs[row][0];
#pragma unroll
    for (int t = 0; t < 4; ++t)
      vals[t][r] = acc[t][r] + brow[t * 16 + nlo];
  }
#pragma unroll
  for (int r = 0; r < 8; ++r) {
    float mx = fmaxf(fmaxf(vals[0][r], vals[1][r]), fmaxf(vals[2][r], vals[3][r]));
    mx = swz_max<SWZ_X1>(mx); mx = swz_max<SWZ_X2>(mx);
    mx = swz_max<SWZ_X4>(mx); mx = swz_max<SWZ_X8>(mx);
    float sm = 0.f;
#pragma unroll
    for (int t = 0; t < 4; ++t) { vals[t][r] = __expf(vals[t][r] - mx); sm += vals[t][r]; }
    sm = swz_add<SWZ_X1>(sm); sm = swz_add<SWZ_X2>(sm);
    sm = swz_add<SWZ_X4>(sm); sm = swz_add<SWZ_X8>(sm);
    const float inv = 1.0f / sm;
    const int row = m0 + half * 8 + r;
#pragma unroll
    for (int t = 0; t < 4; ++t)
      ps[row][t * 16 + nlo] = (bf16)(vals[t][r] * inv);
  }
  __syncthreads();

  // out = P @ V : A = P rows (K = 64 keys), Bt[n=d][k=key] = vts
  v8f o[2] = {zero, zero};
#pragma unroll
  for (int kk = 0; kk < NP; kk += 32) {
    v16bf pa = load_a_frag(&ps[0][0], 72, m0, kk, lane);
#pragma unroll
    for (int t = 0; t < 2; ++t) {
      v16bf bv = load_bt_frag(&vts[0][0], 72, kk, t * 16, lane);
      o[t] = wmma_bf16(pa, bv, o[t]);
    }
  }
  // scatter to token layout
#pragma unroll
  for (int r = 0; r < 8; ++r) {
    const int row = m0 + half * 8 + r;
    if (row < Nq) {
      bf16* orow = outp + (size_t)toks[row] * Cq + head * HD;
#pragma unroll
      for (int t = 0; t < 2; ++t) orow[t * 16 + nlo] = (bf16)o[t][r];
    }
  }
}

// ---------------------------------------------------------------------------
// Launch
// ---------------------------------------------------------------------------
extern "C" void kernel_launch(void* const* d_in, const int* in_sizes, int n_in,
                              void* d_out, int out_size, void* d_ws, size_t ws_size,
                              hipStream_t stream) {
  const float* x        = (const float*)d_in[0];
  const float* attn_msk = (const float*)d_in[1];
  const float* n1g      = (const float*)d_in[2];
  const float* n1b      = (const float*)d_in[3];
  const float* qkv_w    = (const float*)d_in[4];
  const float* qkv_b    = (const float*)d_in[5];
  const float* rpb      = (const float*)d_in[6];
  const float* proj_w   = (const float*)d_in[7];
  const float* proj_b   = (const float*)d_in[8];
  const float* n2g      = (const float*)d_in[9];
  const float* n2b      = (const float*)d_in[10];
  const float* fc1_w    = (const float*)d_in[11];
  const float* fc1_b    = (const float*)d_in[12];
  const float* fc2_w    = (const float*)d_in[13];
  const float* fc2_b    = (const float*)d_in[14];
  float* out = (float*)d_out;

  // Workspace layout (256-aligned regions, reused across phases):
  char* p = (char*)d_ws;
  auto carve = [&](size_t bytes) { char* q = p; p += (bytes + 255) & ~(size_t)255; return q; };
  bf16* wq  = (bf16*)carve((size_t)Cq * 3 * Cq * 2);
  bf16* wp  = (bf16*)carve((size_t)Cq * Cq * 2);
  bf16* w1  = (bf16*)carve((size_t)Cq * 4 * Cq * 2);
  bf16* w2  = (bf16*)carve((size_t)4 * Cq * Cq * 2);
  // Region A: qkv activations (57.8 MB), later fc1 output (77.1 MB)
  char* regA = carve((size_t)MTOK * 4 * Cq * 2);
  bf16* qkv_act = (bf16*)regA;                 // [MTOK, 1152]
  bf16* fc1_out = (bf16*)regA;                 // [MTOK, 1536]
  // Region B: ln1 out -> attn out -> ln2 out (19.3 MB, token layout)
  char* regB = carve((size_t)MTOK * Cq * 2);
  bf16* xln      = (bf16*)regB;
  bf16* attn_out = (bf16*)regB;
  bf16* h2       = (bf16*)regB;
  // Region C: x1 f32 (38.5 MB)
  float* x1 = (float*)carve((size_t)MTOK * Cq * 4);

  // 1) weights -> bf16
  {
    int n;
    n = Cq * 3 * Cq;   cvt_bf16<<<(n + 255) / 256, 256, 0, stream>>>(qkv_w, wq, n);
    n = Cq * Cq;       cvt_bf16<<<(n + 255) / 256, 256, 0, stream>>>(proj_w, wp, n);
    n = Cq * 4 * Cq;   cvt_bf16<<<(n + 255) / 256, 256, 0, stream>>>(fc1_w, w1, n);
    n = 4 * Cq * Cq;   cvt_bf16<<<(n + 255) / 256, 256, 0, stream>>>(fc2_w, w2, n);
  }
  // 2) LN1 (token order; window gather happens inside attention)
  ln_rows<<<MTOK, 128, 0, stream>>>(x, n1g, n1b, xln);
  // 3) QKV GEMM [25088x384]x[384x1152]
  gemm_bf16<EPI_BF16><<<dim3(3 * Cq / 64, MTOK / 256), 256, 0, stream>>>(
      xln, wq, qkv_b, MTOK, 3 * Cq, Cq, qkv_act, nullptr, nullptr);
  // 4) windowed attention (gather/scatter token layout)
  attn_window<<<NWIN * HEADS, 128, 0, stream>>>(qkv_act, rpb, attn_msk, attn_out);
  // 5) proj GEMM + residual -> x1
  gemm_bf16<EPI_RES><<<dim3(Cq / 64, MTOK / 256), 256, 0, stream>>>(
      attn_out, wp, proj_b, MTOK, Cq, Cq, nullptr, x1, x);
  // 6) LN2
  ln_rows<<<MTOK, 128, 0, stream>>>(x1, n2g, n2b, h2);
  // 7) FC1 + GELU
  gemm_bf16<EPI_GELU><<<dim3(4 * Cq / 64, MTOK / 256), 256, 0, stream>>>(
      h2, w1, fc1_b, MTOK, 4 * Cq, Cq, fc1_out, nullptr, nullptr);
  // 8) FC2 + residual -> out
  gemm_bf16<EPI_RES><<<dim3(Cq / 64, MTOK / 256), 256, 0, stream>>>(
      fc1_out, w2, fc2_b, MTOK, Cq, 4 * Cq, nullptr, out, x1);
}
